// TurboQuantProd_59571196395523
// MI455X (gfx1250) — matrix operand order repaired and checked by reading, not compiled, not run on hardware
//
#include <hip/hip_runtime.h>
#include <math.h>

// ---------------------------------------------------------------------------
// TurboQuantProd fused quantize+dequantize for MI455X (gfx1250, wave32, WMMA)
//
// Outputs (concatenated flat in d_out, all as float):
//   x_hat   [T,128]   x_mse + scale*||r||*(sign(r @ P^T) @ P)
//   idx     [T,128]   searchsorted index (stored as float)
//   signs   [T,128]   sign(r @ P^T) in {-1,0,+1}
//   norm    [T]       ||r||_2
// ---------------------------------------------------------------------------

typedef __attribute__((ext_vector_type(16))) _Float16 v16h;
typedef __attribute__((ext_vector_type(8)))  _Float16 v8h;
typedef __attribute__((ext_vector_type(8)))  float    v8f;

#define DIM 128
#define WAVES 4
#define TOKS_PER_WAVE 16
#define TOKS_PER_WG (WAVES * TOKS_PER_WAVE)

// Offset (in halves) of element (k,n) of a K=128 x N=128 B-matrix stored as
// WMMA 16x16x32 f16 B fragments: frag tile (nt,kt) is 32 lanes x 16 halves.
// B layout (per ISA 05_wmma): lane = n%16 + 16*((k%16)>=8), half = k%8 + 8*((k%32)>=16).
__device__ __forceinline__ int bfrag_off(int k, int n) {
  const int kt = k >> 5, kk = k & 31, nt = n >> 4, nn = n & 15;
  const int lane = nn + ((kk & 8) << 1);
  return (((nt << 2) + kt) << 9) + (lane << 4) + ((kk & 7) | ((kk & 16) >> 1));
}

// ---------------------------------------------------------------------------
// Prep: swizzle projection P (128x128 f32, row-major P[e][d]) into f16 WMMA
// B-fragment tables in workspace:
//   bf1: B1[k=d][n=e] = P[e][d]   (for proj_r = r @ P^T)
//   bf2: B2[k=e][n=d] = P[e][d]   (for x_qjl  = signs @ P)
// ---------------------------------------------------------------------------
__global__ void tq_prep(const float* __restrict__ proj,
                        _Float16* __restrict__ bf1,
                        _Float16* __restrict__ bf2) {
  const int i = blockIdx.x * blockDim.x + threadIdx.x;
  if (i < DIM * DIM) {
    const int e = i >> 7, d = i & 127;
    const _Float16 v = (_Float16)proj[i];
    bf1[bfrag_off(d, e)] = v;
    bf2[bfrag_off(e, d)] = v;
  }
}

// ---------------------------------------------------------------------------
// Main fused kernel: 4 waves/WG, 16 tokens per wave.
// ---------------------------------------------------------------------------
__global__ void __launch_bounds__(WAVES * 32)
tq_main(const float* __restrict__ x, const float* __restrict__ cbp,
        const _Float16* __restrict__ bf1, const _Float16* __restrict__ bf2,
        float* __restrict__ out_xhat, float* __restrict__ out_idx,
        float* __restrict__ out_sign, float* __restrict__ out_norm,
        int ntok) {
  __shared__ __attribute__((aligned(16))) _Float16 xmbuf[WAVES][TOKS_PER_WAVE * DIM];
  __shared__ __attribute__((aligned(16))) _Float16 sbuf [WAVES][TOKS_PER_WAVE * DIM];
  __shared__ float normbuf[WAVES][TOKS_PER_WAVE];

  const int tid  = threadIdx.x;
  const int wave = tid >> 5;
  const int lane = tid & 31;
  const int m    = lane & 15;   // A/C-layout row within 16-token tile
  const int hl   = lane >> 4;   // lane half

  const int blockTok = blockIdx.x * TOKS_PER_WG;
  if (blockTok >= ntok) return;
  const int tokBase = blockTok + wave * TOKS_PER_WAVE;

  // codebook + searchsorted boundaries (wave-uniform scalars)
  const float c0 = cbp[0], c1 = cbp[1], c2 = cbp[2], c3 = cbp[3];
  const float bb0 = 0.5f * (c0 + c1);
  const float bb1 = 0.5f * (c1 + c2);
  const float bb2 = 0.5f * (c2 + c3);

  // Pull the fragment tables toward WGP$/L2 (gfx1250 global_prefetch_b8).
  __builtin_prefetch(bf1 + (tid << 7), 0, 1);
  __builtin_prefetch(bf2 + (tid << 7), 0, 1);

  // ---- Phase A: quantize, residual directly in A-fragment register layout.
  // 16-bit A 16x32 layout: lane half hl owns K = kt*32 + {hl*8..hl*8+7} in
  // halves 0..7 and K = kt*32 + 16 + {hl*8..} in halves 8..15.
  const long rowOff = (long)(tokBase + m) * DIM;
  const float* xrow = x + rowOff;
  float* idxrow = out_idx + rowOff;

  v16h afrag[4];
  float sq = 0.f;
#pragma unroll
  for (int kt = 0; kt < 4; ++kt) {
#pragma unroll
    for (int run = 0; run < 2; ++run) {
      const int d0 = kt * 32 + run * 16 + hl * 8;
      const float4 p0 = *(const float4*)(xrow + d0);
      const float4 p1 = *(const float4*)(xrow + d0 + 4);
      const float xv[8] = {p0.x, p0.y, p0.z, p0.w, p1.x, p1.y, p1.z, p1.w};
      v8h xm8;
      float iv[8];
#pragma unroll
      for (int j = 0; j < 8; ++j) {
        const float v = xv[j];
        const int idx = (int)(v > bb0) + (int)(v > bb1) + (int)(v > bb2);
        const float xm =
            (idx & 2) ? ((idx & 1) ? c3 : c2) : ((idx & 1) ? c1 : c0);
        const float r = v - xm;
        sq += r * r;
        afrag[kt][run * 8 + j] = (_Float16)r;
        xm8[j] = (_Float16)xm;
        iv[j] = (float)idx;
      }
      *(float4*)(idxrow + d0)     = make_float4(iv[0], iv[1], iv[2], iv[3]);
      *(float4*)(idxrow + d0 + 4) = make_float4(iv[4], iv[5], iv[6], iv[7]);
      *(v8h*)(&xmbuf[wave][m * DIM + d0]) = xm8;  // ds_store_b128
    }
  }

  // Residual L2 norm: lanes l and l^16 hold complementary halves of row m.
  const float nrm = sqrtf(sq + __shfl_xor(sq, 16, 32));
  if (lane < 16) {
    out_norm[tokBase + m] = nrm;
    normbuf[wave][m] = nrm;
  }

  // ---- Phase B: proj_r = r @ P^T  (8 N-tiles x 4 K-steps of WMMA f16).
  v8f acc[8];
#pragma unroll
  for (int nt = 0; nt < 8; ++nt) {
    v8f c = {0.f, 0.f, 0.f, 0.f, 0.f, 0.f, 0.f, 0.f};
#pragma unroll
    for (int kt = 0; kt < 4; ++kt) {
      const v16h b =
          *(const v16h*)(bf1 + ((((nt << 2) + kt) << 9) + (lane << 4)));
      c = __builtin_amdgcn_wmma_f32_16x16x32_f16(false, afrag[kt], false, b,
                                                 (short)0, c, false, false);
    }
    acc[nt] = c;
  }

  // ---- Signs: C layout (lane<16: M=v, lane>=16: M=v+8; N = nt*16 + lane%16)
#pragma unroll
  for (int nt = 0; nt < 8; ++nt) {
#pragma unroll
    for (int v = 0; v < 8; ++v) {
      const float p = acc[nt][v];
      const float s = (p > 0.f) ? 1.f : ((p < 0.f) ? -1.f : 0.f);
      const int row = v + (hl << 3);
      const int col = (nt << 4) + m;
      out_sign[(long)(tokBase + row) * DIM + col] = s;
      sbuf[wave][row * DIM + col] = (_Float16)s;
    }
  }
  __syncthreads();  // make sbuf / xmbuf / normbuf cross-lane visible

  // ---- Phase C: gather signs into A-fragment layout, x_qjl = signs @ P.
  v16h sfrag[4];
#pragma unroll
  for (int kt = 0; kt < 4; ++kt) {
    const int d0 = kt * 32 + hl * 8;
    const v8h lo = *(const v8h*)(&sbuf[wave][m * DIM + d0]);       // ds_load_b128
    const v8h hi = *(const v8h*)(&sbuf[wave][m * DIM + d0 + 16]);  // ds_load_b128
#pragma unroll
    for (int j = 0; j < 8; ++j) {
      sfrag[kt][j]     = lo[j];
      sfrag[kt][j + 8] = hi[j];
    }
  }
#pragma unroll
  for (int nt = 0; nt < 8; ++nt) {
    v8f c = {0.f, 0.f, 0.f, 0.f, 0.f, 0.f, 0.f, 0.f};
#pragma unroll
    for (int kt = 0; kt < 4; ++kt) {
      const v16h b =
          *(const v16h*)(bf2 + ((((nt << 2) + kt) << 9) + (lane << 4)));
      c = __builtin_amdgcn_wmma_f32_16x16x32_f16(false, sfrag[kt], false, b,
                                                 (short)0, c, false, false);
    }
    acc[nt] = c;
  }

  // ---- Phase D: x_hat = x_mse + (sqrt(pi/2)/128) * ||r|| * x_qjl
  const float scale = 0.009791516697f;  // sqrt(pi/2) / 128
  float nrm8[8];
#pragma unroll
  for (int v = 0; v < 8; ++v) nrm8[v] = normbuf[wave][v + (hl << 3)] * scale;
#pragma unroll
  for (int nt = 0; nt < 8; ++nt) {
#pragma unroll
    for (int v = 0; v < 8; ++v) {
      const int row = v + (hl << 3);
      const int col = (nt << 4) + m;
      const float xm = (float)xmbuf[wave][row * DIM + col];
      out_xhat[(long)(tokBase + row) * DIM + col] = xm + nrm8[v] * acc[nt][v];
    }
  }
}

// ---------------------------------------------------------------------------
extern "C" void kernel_launch(void* const* d_in, const int* in_sizes, int n_in,
                              void* d_out, int out_size, void* d_ws,
                              size_t ws_size, hipStream_t stream) {
  const float* x    = (const float*)d_in[0];  // [B,H,S,128] f32
  const float* cb   = (const float*)d_in[1];  // [4] f32
  const float* proj = (const float*)d_in[2];  // [128,128] f32

  const int total = in_sizes[0];        // B*H*S*128
  const int ntok  = total >> 7;         // B*H*S

  float* out_xhat = (float*)d_out;
  float* out_idx  = out_xhat + (size_t)total;
  float* out_sign = out_idx + (size_t)total;
  float* out_norm = out_sign + (size_t)total;

  // Workspace: two 32KB f16 fragment tables (64KB total), L2-resident.
  _Float16* bf1 = (_Float16*)d_ws;
  _Float16* bf2 = bf1 + DIM * DIM;

  tq_prep<<<(DIM * DIM + 255) / 256, 256, 0, stream>>>(proj, bf1, bf2);

  const int grid = (ntok + TOKS_PER_WG - 1) / TOKS_PER_WG;
  tq_main<<<grid, WAVES * 32, 0, stream>>>(x, cb, bf1, bf2, out_xhat, out_idx,
                                           out_sign, out_norm, ntok);
}
